// AliNetGraphAttentionLayer_30872224923722
// MI455X (gfx1250) — matrix operand order repaired and checked by reading, not compiled, not run on hardware
//
#include <hip/hip_runtime.h>
#include <hip/hip_bf16.h>

typedef __attribute__((ext_vector_type(16))) _Float16 v16h;
typedef __attribute__((ext_vector_type(8)))  float    v8f;

#define D 128
#define LDSW 136   // padded row stride (halves): 272B = 16*17 -> 16B-aligned rows,
                   // 68-dword stride puts the 16 fragment lanes on distinct banks

// ---------------------------------------------------------------- fill utils
__global__ void fill_f32(float* __restrict__ p, float v, long long n) {
    long long i = (long long)blockIdx.x * blockDim.x + threadIdx.x;
    long long stride = (long long)gridDim.x * blockDim.x;
    for (; i < n; i += stride) p[i] = v;
}

// ---------------------------------------------------------------- BN stats
__global__ void bn_stats(const float* __restrict__ x, float* __restrict__ sum,
                         float* __restrict__ sumsq, int N) {
    int f = threadIdx.x;
    float s = 0.f, sq = 0.f;
    for (int r = blockIdx.x; r < N; r += gridDim.x) {
        float v = x[(long long)r * D + f];
        s += v; sq += v * v;
    }
    atomicAdd(&sum[f], s);
    atomicAdd(&sumsq[f], sq);
}

__global__ void bn_finalize(const float* __restrict__ sum, const float* __restrict__ sumsq,
                            const float* __restrict__ gamma, const float* __restrict__ beta,
                            float* __restrict__ scale, float* __restrict__ shift, int N) {
    int f = threadIdx.x;
    float inv_n = 1.0f / (float)N;
    float mean = sum[f] * inv_n;
    float var  = sumsq[f] * inv_n - mean * mean;   // biased variance (matches jnp.var)
    float sc   = gamma[f] * rsqrtf(var + 1e-5f);
    scale[f] = sc;
    shift[f] = beta[f] - mean * sc;
}

// ---------------------------------------------------------------- weight pre-pack
// One-shot: w[m][n][k] = (f16)Km[k][n]  (transposed, f16) -> shared by all GEMM blocks.
__global__ void pack_weights(const float* __restrict__ K0, const float* __restrict__ K1,
                             const float* __restrict__ K2, _Float16* __restrict__ w) {
    int i = blockIdx.x * blockDim.x + threadIdx.x;   // 3*128*128 threads total
    if (i >= 3 * D * D) return;
    int m = i / (D * D), r = i % (D * D);
    int n = r >> 7, k = r & 127;
    const float* Km = (m == 0) ? K0 : (m == 1) ? K1 : K2;
    w[i] = (_Float16)Km[k * D + n];
}

// ---------------------------------------------------------------- fused BN + 3 GEMMs (WMMA)
// Block = 256 threads = 8 waves; block owns 128 rows. xn (f16) staged via BN-affine
// convert; packed f16 weights staged via CDNA5 async global->LDS (ASYNCcnt) copies.
// Each wave: v_wmma_f32_16x16x32_f16 over 4 K-tiles x 8 N-tiles x 3 matrices.
__global__ __launch_bounds__(256)
void fused_bn_gemm(const float* __restrict__ x,
                   const float* __restrict__ scale, const float* __restrict__ shift,
                   const _Float16* __restrict__ w,
                   float* __restrict__ mapped, float* __restrict__ s1,
                   float* __restrict__ s2, int N) {
    extern __shared__ _Float16 smem[];
    _Float16* xnS = smem;              // [128][LDSW] (row, k)
    _Float16* ktS = smem + D * LDSW;   // [128][LDSW] (n, k) transposed weights
    const unsigned ktS_off = (unsigned)(D * LDSW * sizeof(_Float16)); // LDS byte offset

    const int t    = threadIdx.x;
    const int lane = t & 31;
    const int wave = t >> 5;
    const int half = lane >> 4;
    const int ln   = lane & 15;
    const int rowTile = wave * 16;
    const long long blockRow0 = (long long)blockIdx.x * D;

    // Stage normalized x tile (f32 -> BN affine -> f16) in LDS.
    for (int i = t; i < D * D; i += 256) {
        int r = i >> 7, c = i & 127;
        long long gr = blockRow0 + r;
        float v = (gr < N) ? x[gr * D + c] * scale[c] + shift[c] : 0.0f;
        xnS[r * LDSW + c] = (_Float16)v;
    }
    __syncthreads();

    for (int m = 0; m < 3; ++m) {
        // Async global->LDS stage of packed weight matrix m (32KB), B128 chunks.
        // LDS rows padded to LDSW halves; every chunk 16B aligned on both sides.
        {
            unsigned long long wbase = (unsigned long long)(w + (size_t)m * D * D);
            #pragma unroll
            for (int it = 0; it < 8; ++it) {
                int i = t + it * 256;              // 2048 chunks of 16B
                int n = i >> 4, c = i & 15;
                unsigned goff = (unsigned)((n * D + c * 8) * sizeof(_Float16));
                unsigned lds  = ktS_off + (unsigned)((n * LDSW + c * 8) * sizeof(_Float16));
                asm volatile("global_load_async_to_lds_b128 %0, %1, %2"
                             :: "v"(lds), "v"(goff), "s"(wbase) : "memory");
            }
            asm volatile("s_wait_asynccnt 0x0" ::: "memory");
        }
        __syncthreads();

        v8f acc[8];
        #pragma unroll
        for (int nt = 0; nt < 8; ++nt) { v8f z = {}; acc[nt] = z; }

        #pragma unroll
        for (int kk = 0; kk < D; kk += 32) {
            // A fragment (16x32 f16): lane ln = row M; VGPR pair j holds
            // K = (j<4 ? 2j : 16+2(j-4)) + 8*half (+kk).
            v16h a;
            const _Float16* xr = &xnS[(rowTile + ln) * LDSW + kk + 8 * half];
            #pragma unroll
            for (int j = 0; j < 8; ++j) {
                int kb = (j < 4) ? (2 * j) : (16 + 2 * (j - 4));
                a[2 * j]     = xr[kb];
                a[2 * j + 1] = xr[kb + 1];
            }
            #pragma unroll
            for (int nt = 0; nt < 8; ++nt) {
                // B fragment (32x16 f16): lane ln = column N, element e -> K = (e<8 ? e : e+8) + 8*half.
                v16h b;
                const _Float16* kc = &ktS[(nt * 16 + ln) * LDSW + kk + 8 * half];
                #pragma unroll
                for (int e = 0; e < 16; ++e) b[e] = kc[(e < 8) ? e : (e + 8)];
                acc[nt] = __builtin_amdgcn_wmma_f32_16x16x32_f16(
                    false, a, false, b, (short)0, acc[nt], false, false);
            }
        }

        if (m == 0) {
            // C/D layout: acc[nt][v] = (row = rowTile + v + 8*half, col = nt*16 + ln)
            #pragma unroll
            for (int nt = 0; nt < 8; ++nt)
                #pragma unroll
                for (int v = 0; v < 8; ++v) {
                    long long gr = blockRow0 + rowTile + v + 8 * half;
                    if (gr < N) mapped[gr * D + nt * 16 + ln] = acc[nt][v];
                }
        } else {
            // s = tanh(sum_cols((xn @ Km) * xn)): per-row dot, 16-lane shuffle reduce.
            float rd[8] = {0.f, 0.f, 0.f, 0.f, 0.f, 0.f, 0.f, 0.f};
            #pragma unroll
            for (int nt = 0; nt < 8; ++nt)
                #pragma unroll
                for (int v = 0; v < 8; ++v) {
                    int r = rowTile + v + 8 * half;
                    rd[v] += acc[nt][v] * (float)xnS[r * LDSW + nt * 16 + ln];
                }
            #pragma unroll
            for (int off = 1; off < 16; off <<= 1)
                #pragma unroll
                for (int v = 0; v < 8; ++v) rd[v] += __shfl_xor(rd[v], off, 32);
            if (ln == 0) {
                float* sp = (m == 1) ? s1 : s2;
                #pragma unroll
                for (int v = 0; v < 8; ++v) {
                    long long gr = blockRow0 + rowTile + v + 8 * half;
                    if (gr < N) sp[gr] = tanhf(rd[v]);
                }
            }
        }
        __syncthreads();
    }
}

// ---------------------------------------------------------------- edge pass 1: score + segment max
__global__ void edge_score(const long long* __restrict__ src, const long long* __restrict__ dst,
                           const float* __restrict__ s1, const float* __restrict__ s2,
                           float* __restrict__ ews, float* __restrict__ emax, long long E) {
    long long i = (long long)blockIdx.x * blockDim.x + threadIdx.x;
    if (i >= E) return;
    long long s = src[i], d = dst[i];
    float e = s1[s] + s2[d];
    e = (e > 0.f) ? e : 0.01f * e;           // leaky_relu(0.01)
    ews[i] = e;
    // float atomic max via int-max / uint-min trick (init = -inf)
    if (e >= 0.f) atomicMax((int*)&emax[s], __float_as_int(e));
    else          atomicMin((unsigned int*)&emax[s], __float_as_uint(e));
}

// ---------------------------------------------------------------- edge pass 2: exp + segment sum
__global__ void edge_exp(const long long* __restrict__ src, float* __restrict__ ews,
                         const float* __restrict__ emax, float* __restrict__ denom, long long E) {
    long long i = (long long)blockIdx.x * blockDim.x + threadIdx.x;
    if (i >= E) return;
    long long s = src[i];
    float ee = __expf(ews[i] - emax[s]);     // emax finite for every src with edges
    ews[i] = ee;
    atomicAdd(&denom[s], ee);
}

// ---------------------------------------------------------------- edge pass 3: weighted scatter-add
// One wave per edge; lane handles 4 features (float4 gather from L2-resident mapped,
// 4x global_atomic_add_f32 into out).
__global__ __launch_bounds__(256)
void edge_aggregate(const long long* __restrict__ src, const long long* __restrict__ dst,
                    const float* __restrict__ ews, const float* __restrict__ denom,
                    const float* __restrict__ mapped, float* __restrict__ out, long long E) {
    long long edge = (long long)blockIdx.x * 8 + (threadIdx.x >> 5);
    int lane = threadIdx.x & 31;
    if (edge >= E) return;
    long long s = src[edge], d = dst[edge];
    float a = ews[edge] / fmaxf(denom[s], 1e-16f);
    const float4 mv = *(const float4*)&mapped[d * D + lane * 4];
    float* op = &out[s * D + lane * 4];
    atomicAdd(op + 0, a * mv.x);
    atomicAdd(op + 1, a * mv.y);
    atomicAdd(op + 2, a * mv.z);
    atomicAdd(op + 3, a * mv.w);
}

// ---------------------------------------------------------------- final relu
__global__ void relu_k(float* __restrict__ out, long long n) {
    long long i = (long long)blockIdx.x * blockDim.x + threadIdx.x;
    long long stride = (long long)gridDim.x * blockDim.x;
    for (; i < n; i += stride) out[i] = fmaxf(out[i], 0.f);
}

// ---------------------------------------------------------------- launcher
extern "C" void kernel_launch(void* const* d_in, const int* in_sizes, int n_in,
                              void* d_out, int out_size, void* d_ws, size_t ws_size,
                              hipStream_t stream) {
    const float*     x     = (const float*)d_in[0];
    const long long* ei    = (const long long*)d_in[1];   // int64 [2, E]
    const float*     K0    = (const float*)d_in[2];
    const float*     K1    = (const float*)d_in[3];
    const float*     K2    = (const float*)d_in[4];
    const float*     gamma = (const float*)d_in[5];
    const float*     beta  = (const float*)d_in[6];
    float* out = (float*)d_out;

    const long long N = in_sizes[0] / D;
    const long long E = in_sizes[1] / 2;
    const long long* src = ei;
    const long long* dst = ei + E;

    // workspace carve-up (floats)
    float* ws     = (float*)d_ws;
    float* sum    = ws;               // 128
    float* sumsq  = ws + 128;         // 128
    float* scl    = ws + 256;         // 128
    float* sft    = ws + 384;         // 128
    float* mapped = ws + 512;         // N*128
    float* s1     = mapped + N * D;   // N
    float* s2     = s1 + N;           // N
    float* emax   = s2 + N;           // N
    float* denom  = emax + N;         // N
    float* ews    = denom + N;        // E
    _Float16* wf16 = (_Float16*)(ews + E);  // 3*128*128 f16 (16B-aligned offset)

    // deterministic init (fill kernels: graph-capture safe)
    fill_f32<<<1, 256, 0, stream>>>(sum, 0.f, 256);                         // sum+sumsq
    fill_f32<<<(int)((N + 255) / 256), 256, 0, stream>>>(denom, 0.f, N);
    fill_f32<<<(int)((N + 255) / 256), 256, 0, stream>>>(emax, -__builtin_inff(), N);
    fill_f32<<<2048, 256, 0, stream>>>(out, 0.f, (long long)out_size);      // scatter target

    // phase 1: BN statistics + one-shot weight pack (transpose + f16)
    bn_stats<<<512, 128, 0, stream>>>(x, sum, sumsq, (int)N);
    bn_finalize<<<1, 128, 0, stream>>>(sum, sumsq, gamma, beta, scl, sft, (int)N);
    pack_weights<<<(3 * D * D + 255) / 256, 256, 0, stream>>>(K0, K1, K2, wf16);

    // phase 2: fused BN-apply + 3 WMMA GEMMs (mapped, s1, s2)
    int nBlocks = (int)((N + D - 1) / D);
    size_t shmem = (size_t)2 * D * LDSW * sizeof(_Float16);   // 68 KB < 320 KB/WGP
    fused_bn_gemm<<<nBlocks, 256, shmem, stream>>>(x, scl, sft, wf16,
                                                   mapped, s1, s2, (int)N);

    // phase 3: edge softmax + aggregation (L2-resident gather/scatter)
    int eBlocks = (int)((E + 255) / 256);
    edge_score<<<eBlocks, 256, 0, stream>>>(src, dst, s1, s2, ews, emax, E);
    edge_exp<<<eBlocks, 256, 0, stream>>>(src, ews, emax, denom, E);
    edge_aggregate<<<(int)((E + 7) / 8), 256, 0, stream>>>(src, dst, ews, denom,
                                                           mapped, out, E);
    relu_k<<<2048, 256, 0, stream>>>(out, (long long)out_size);
}
